// SnliModel_33414845563375
// MI455X (gfx1250) — matrix operand order repaired and checked by reading, not compiled, not run on hardware
//
#include <hip/hip_runtime.h>

typedef __attribute__((ext_vector_type(16))) __bf16        v16bf;
typedef __attribute__((ext_vector_type(8)))  float         v8f;
typedef __attribute__((ext_vector_type(4)))  unsigned int  u32x4;
typedef __attribute__((ext_vector_type(8)))  unsigned int  u32x8;

#define HDIM  256
#define EDIM  300
#define EPAD  320
#define BSZ   32
#define LEN   128
#define NSTEP 127
#define AMBN  32
#define NROWS 255      /* LEN + NSTEP */
#define FDIM  1280     /* 5*HDIM */

__device__ __forceinline__ unsigned short f32_bf16(float f) {
  unsigned int u = __builtin_bit_cast(unsigned int, f);
  u += 0x7FFFu + ((u >> 16) & 1u);            // round-to-nearest-even
  return (unsigned short)(u >> 16);
}
__device__ __forceinline__ float sigmoidf_(float x) { return 1.0f / (1.0f + __expf(-x)); }

__device__ __forceinline__ v16bf mk_frag(u32x4 lo, u32x4 hi) {
  u32x8 v;
  v[0]=lo[0]; v[1]=lo[1]; v[2]=lo[2]; v[3]=lo[3];
  v[4]=hi[0]; v[5]=hi[1]; v[6]=hi[2]; v[7]=hi[3];
  return __builtin_bit_cast(v16bf, v);
}
// A fragment (16x32 bf16): lane = (m = lane&15, khalf = lane>>4).
// halfs 0..7  -> k =  8*khalf + h ;  halfs 8..15 -> k = 16 + 8*khalf + (h-8)
__device__ __forceinline__ v16bf load_a_frag(const unsigned short* base, int lane, int ld) {
  const unsigned short* row = base + (lane & 15) * ld + ((lane >> 4) << 3);
  u32x4 lo = *(const u32x4*)(row);
  u32x4 hi = *(const u32x4*)(row + 16);
  return mk_frag(lo, hi);
}
// B fragment (32x16 bf16) with B[k][n] = src[n][k] (src row-major, stride ld):
// lane = (n = lane&15), halfs 0..7 -> k = 8*(lane>>4) + h ; halfs 8..15 -> k = 16 + 8*(lane>>4) + (h-8).
__device__ __forceinline__ v16bf load_b_frag(const unsigned short* base, int lane, int ld) {
  const unsigned short* p = base + (lane & 15) * ld + ((lane >> 4) << 3);
  u32x4 lo = *(const u32x4*)(p);
  u32x4 hi = *(const u32x4*)(p + 16);
  return mk_frag(lo, hi);
}

// ---------------- prep: f32 -> bf16 weight copies ----------------
__global__ __launch_bounds__(256) void k_prep(const float* __restrict__ W,
                                              const float* __restrict__ U,
                                              unsigned short* __restrict__ Wb,
                                              unsigned short* __restrict__ Ub) {
  long i = (long)blockIdx.x * 256 + threadIdx.x;
  long stride = (long)gridDim.x * 256;
  long nW = (long)FDIM * EPAD;
  for (long idx = i; idx < nW; idx += stride) {
    int row = (int)(idx / EPAD), k = (int)(idx % EPAD);
    float v = (k < EDIM) ? W[(long)row * EDIM + k] : 0.0f;   // zero-pad K 300->320
    Wb[idx] = f32_bf16(v);
  }
  long nU = (long)FDIM * (2 * HDIM);
  for (long idx = i; idx < nU; idx += stride) Ub[idx] = f32_bf16(U[idx]);
}

__global__ __launch_bounds__(256) void k_unorm(const float* __restrict__ eu, float* __restrict__ out) {
  __shared__ float red[256];
  int t = threadIdx.x;
  float v = eu[t];                 // HDIM == 256, one element per thread
  red[t] = v * v;
  __syncthreads();
  for (int s = 128; s > 0; s >>= 1) { if (t < s) red[t] += red[t + s]; __syncthreads(); }
  if (t == 0) *out = sqrtf(red[0]);
}

// ---------------- leaves: pre = emb @ W^T + b ; c0,h0 -> chart ----------------
__global__ __launch_bounds__(256) void k_leaves(const unsigned short* __restrict__ Wb,
                                                const float* __restrict__ bias,
                                                const float* __restrict__ wemb,
                                                const int* __restrict__ sent1,
                                                const int* __restrict__ sent2,
                                                float* __restrict__ chart) {
  const int snt = blockIdx.y;
  const int mtile = blockIdx.x;          // 0..255 : (b, 16-token tile)
  const int b  = mtile >> 3;
  const int lt = mtile & 7;
  const int* toks = (snt ? sent2 : sent1) + b * LEN + lt * 16;

  __shared__ unsigned short Ab[16][EPAD];          // 10 KB bf16 A tile
  const int tid = threadIdx.x;
  for (int idx = tid; idx < 16 * EPAD; idx += 256) {
    int mr = idx / EPAD, k = idx % EPAD;
    float v = (k < EDIM) ? wemb[(long)toks[mr] * EDIM + k] : 0.0f;
    Ab[mr][k] = f32_bf16(v);
  }
  __syncthreads();

  const int wave = tid >> 5, lane = tid & 31;
  float* crow_base = chart + ((long)(snt * BSZ + b) * NROWS + (long)lt * 16) * (2 * HDIM);
  const int gates[3] = {0, 3, 4};                  // only i, o, u needed at leaves

  for (int hi = 0; hi < 2; ++hi) {
    int ht = wave * 2 + hi;                        // hidden-tile 0..15
    v8f acc[3];
    for (int gi = 0; gi < 3; ++gi) {
      v8f a_ = {0,0,0,0,0,0,0,0};
      int nt = gates[gi] * 16 + ht;
      const unsigned short* Bb = Wb + (long)(nt * 16) * EPAD;
      for (int ks = 0; ks < 10; ++ks) {            // K = 320
        v16bf af = load_a_frag(&Ab[0][0] + ks * 32, lane, EPAD);
        v16bf bf = load_b_frag(Bb + ks * 32, lane, EPAD);
        a_ = __builtin_amdgcn_wmma_f32_16x16x32_bf16(false, af, false, bf, (short)0, a_, false, false);
      }
      acc[gi] = a_;
    }
    int n = ht * 16 + (lane & 15);
    float bi = bias[n], bo = bias[3 * HDIM + n], bu = bias[4 * HDIM + n];
    int mrb = (lane >> 4) * 8;
    for (int r = 0; r < 8; ++r) {
      int m = mrb + r;
      float c0 = sigmoidf_(acc[0][r] + bi) * tanhf(acc[2][r] + bu);
      float h0 = sigmoidf_(acc[1][r] + bo) * tanhf(c0);
      float* row = crow_base + (long)m * (2 * HDIM);
      row[n] = c0;
      row[HDIM + n] = h0;
    }
  }
}

// ---------------- scan: 127 sequential chart-building steps ----------------
__global__ __launch_bounds__(256) void k_scan(const unsigned short* __restrict__ Ub,
                                              const float* __restrict__ bias,
                                              const float* __restrict__ eu,
                                              const float* __restrict__ unormp,
                                              const int* __restrict__ ops1,
                                              const int* __restrict__ ops2,
                                              float* __restrict__ chart) {
  const int snt = blockIdx.y, b = blockIdx.x;
  const int* ops = (snt ? ops2 : ops1);
  float* chrow0 = chart + (long)(snt * BSZ + b) * NROWS * (2 * HDIM);

  __shared__ unsigned short HHs[AMBN][2 * HDIM];   // 32 KB bf16 A operand
  __shared__ float Cms[AMBN][HDIM];                // 32 KB candidate c
  __shared__ float Hms[AMBN][HDIM];                // 32 KB candidate h
  __shared__ float eus[HDIM];                      // staged energy_u
  __shared__ float rnum[AMBN][8], rss[AMBN][8];    // reduction partials
  __shared__ int   opsh[AMBN][2];
  __shared__ float Ev[AMBN], Sw[AMBN];

  const int tid = threadIdx.x, wave = tid >> 5, lane = tid & 31;
  const float unorm = *unormp;
  eus[tid] = eu[tid];                              // HDIM == blockDim == 256

  for (int step = 0; step < NSTEP; ++step) {
    const int* opst = ops + ((long)b * NSTEP + step) * AMBN * 2;
    if (tid < AMBN * 2) ((int*)opsh)[tid] = opst[tid];
    __syncthreads();

    // hh[a] = [h(left) | h(right)]  (bf16)
    for (int idx = tid; idx < AMBN * 2 * HDIM; idx += 256) {
      int a = idx >> 9, k = idx & 511;
      int child = k >> 8, kk = k & 255;
      int rowi = opsh[a][child];
      float v = chrow0[(long)rowi * (2 * HDIM) + HDIM + kk];
      HHs[a][k] = f32_bf16(v);
    }
    __syncthreads();

    // GEMM (32x1280x512): each wave owns 2 hidden-tiles; per k-step one B-frag
    // feeds BOTH m-tiles (2 WMMAs per 32B B load).
    for (int hh = 0; hh < 2; ++hh) {
      int ht = wave * 2 + hh;                      // hidden-tile 0..15
      v8f acc[5][2];
      for (int g = 0; g < 5; ++g)
        for (int mt = 0; mt < 2; ++mt)
          acc[g][mt] = (v8f){0,0,0,0,0,0,0,0};
      for (int ks = 0; ks < 16; ++ks) {            // K = 512
        v16bf a0 = load_a_frag(&HHs[0][0]  + ks * 32, lane, 2 * HDIM);
        v16bf a1 = load_a_frag(&HHs[16][0] + ks * 32, lane, 2 * HDIM);
        for (int g = 0; g < 5; ++g) {
          const unsigned short* Bb = Ub + (long)((g * 16 + ht) * 16) * (2 * HDIM) + ks * 32;
          v16bf bf = load_b_frag(Bb, lane, 2 * HDIM);
          acc[g][0] = __builtin_amdgcn_wmma_f32_16x16x32_bf16(false, a0, false, bf, (short)0, acc[g][0], false, false);
          acc[g][1] = __builtin_amdgcn_wmma_f32_16x16x32_bf16(false, a1, false, bf, (short)0, acc[g][1], false, false);
        }
      }
      // gated epilogue -> LDS candidate matrices
      int n = ht * 16 + (lane & 15);
      float bi = bias[n], bfl = bias[HDIM + n], bfr = bias[2 * HDIM + n];
      float bo = bias[3 * HDIM + n], bu = bias[4 * HDIM + n];
      for (int mt = 0; mt < 2; ++mt) {
        for (int r = 0; r < 8; ++r) {
          int a = mt * 16 + (lane >> 4) * 8 + r;
          float ccl = chrow0[(long)opsh[a][0] * (2 * HDIM) + n];
          float ccr = chrow0[(long)opsh[a][1] * (2 * HDIM) + n];
          float ig = sigmoidf_(acc[0][mt][r] + bi);
          float fl = sigmoidf_(acc[1][mt][r] + bfl);
          float fr = sigmoidf_(acc[2][mt][r] + bfr);
          float og = sigmoidf_(acc[3][mt][r] + bo);
          float ug = tanhf(acc[4][mt][r] + bu);
          float c = fl * ccl + fr * ccr + ig * ug;
          float h = og * tanhf(c);
          Cms[a][n] = c;
          Hms[a][n] = h;
        }
      }
    }
    __syncthreads();

    // cosine energy: 8 threads per candidate, 32-element chunks
    {
      int a = tid >> 3, ch = tid & 7;
      float num = 0.0f, ss = 0.0f;
      int k0 = ch * 32;
      for (int k = k0; k < k0 + 32; ++k) {
        float hv = Hms[a][k];
        num += hv * eus[k];
        ss  += hv * hv;
      }
      rnum[a][ch] = num;
      rss[a][ch]  = ss;
    }
    __syncthreads();
    if (tid < AMBN) {
      float num = 0.0f, ss = 0.0f;
      for (int ch = 0; ch < 8; ++ch) { num += rnum[tid][ch]; ss += rss[tid][ch]; }
      Ev[tid] = num / fmaxf(sqrtf(ss) * unorm, 1e-8f);
    }
    __syncthreads();
    if (tid == 0) {                                // softmax over 32 candidates
      float mx = -1e30f;
      for (int a = 0; a < AMBN; ++a) mx = fmaxf(mx, Ev[a]);
      float sum = 0.0f;
      for (int a = 0; a < AMBN; ++a) { float e = __expf(Ev[a] - mx); Sw[a] = e; sum += e; }
      float inv = 1.0f / sum;
      for (int a = 0; a < AMBN; ++a) Sw[a] *= inv;
    }
    __syncthreads();

    // weighted combine -> new chart row
    {
      float o0 = 0.0f, o1 = 0.0f;
      int f0 = tid;                                // 0..255
      for (int a = 0; a < AMBN; ++a) {
        float s = Sw[a];
        o0 += s * Cms[a][f0];
        o1 += s * Hms[a][f0];
      }
      float* outr = chrow0 + (long)(LEN + step) * (2 * HDIM);
      outr[f0] = o0;
      outr[HDIM + f0] = o1;
    }
    __syncthreads();
  }
}

// ---------------- classifier ----------------
__global__ __launch_bounds__(256) void k_mlp1(const float* __restrict__ chart,
                                              const float* __restrict__ w1,
                                              const float* __restrict__ b1,
                                              float* __restrict__ inter) {
  const int b = blockIdx.x;
  const float* s1 = chart + ((long)(0 * BSZ + b) * NROWS + (NROWS - 1)) * (2 * HDIM) + HDIM;
  const float* s2 = chart + ((long)(1 * BSZ + b) * NROWS + (NROWS - 1)) * (2 * HDIM) + HDIM;
  __shared__ float conc[512];
  const int t = threadIdx.x;
  conc[t] = s1[t];
  conc[256 + t] = s2[t];
  __syncthreads();
  for (int j = t; j < 1024; j += 256) {
    const float* wr = w1 + (long)j * 512;
    float acc = b1[j];
    for (int k = 0; k < 512; ++k) acc += wr[k] * conc[k];
    inter[(long)b * 1024 + j] = fmaxf(acc, 0.0f);
  }
}

__global__ __launch_bounds__(128) void k_mlp2(const float* __restrict__ inter,
                                              const float* __restrict__ w2,
                                              const float* __restrict__ b2,
                                              float* __restrict__ out) {
  __shared__ float lg[BSZ][3];
  const int t = threadIdx.x;
  if (t < BSZ * 3) {
    int b = t / 3, c = t % 3;
    const float* wr = w2 + (long)c * 1024;
    const float* ir = inter + (long)b * 1024;
    float acc = b2[c];
    for (int k = 0; k < 1024; ++k) acc += wr[k] * ir[k];
    lg[b][c] = acc;
  }
  __syncthreads();
  if (t < BSZ) {
    float m = fmaxf(lg[t][0], fmaxf(lg[t][1], lg[t][2]));
    float e0 = __expf(lg[t][0] - m), e1 = __expf(lg[t][1] - m), e2 = __expf(lg[t][2] - m);
    float inv = 1.0f / (e0 + e1 + e2);
    out[t * 3 + 0] = e0 * inv;
    out[t * 3 + 1] = e1 * inv;
    out[t * 3 + 2] = e2 * inv;
  }
}

extern "C" void kernel_launch(void* const* d_in, const int* in_sizes, int n_in,
                              void* d_out, int out_size, void* d_ws, size_t ws_size,
                              hipStream_t stream) {
  (void)in_sizes; (void)n_in; (void)out_size; (void)ws_size;
  const float* W    = (const float*)d_in[0];
  const float* U    = (const float*)d_in[1];
  const float* bv   = (const float*)d_in[2];
  const float* eu   = (const float*)d_in[3];
  const float* wemb = (const float*)d_in[4];
  const float* w1   = (const float*)d_in[5];
  const float* b1   = (const float*)d_in[6];
  const float* w2   = (const float*)d_in[7];
  const float* b2   = (const float*)d_in[8];
  const int* sent1  = (const int*)d_in[9];
  const int* ops1   = (const int*)d_in[10];
  const int* sent2  = (const int*)d_in[11];
  const int* ops2   = (const int*)d_in[12];
  float* out = (float*)d_out;

  char* ws = (char*)d_ws;
  size_t off = 0;
  auto wsalloc = [&](size_t bytes) -> void* {
    void* p = ws + off;
    off = (off + bytes + 255) & ~(size_t)255;
    return p;
  };
  unsigned short* Wb   = (unsigned short*)wsalloc((size_t)FDIM * EPAD * 2);
  unsigned short* Ubw  = (unsigned short*)wsalloc((size_t)FDIM * 2 * HDIM * 2);
  float* unorm         = (float*)wsalloc(4);
  float* chart         = (float*)wsalloc((size_t)2 * BSZ * NROWS * 2 * HDIM * 4);
  float* inter         = (float*)wsalloc((size_t)BSZ * 1024 * 4);

  k_prep  <<<dim3(512),      dim3(256), 0, stream>>>(W, U, Wb, Ubw);
  k_unorm <<<dim3(1),        dim3(256), 0, stream>>>(eu, unorm);
  k_leaves<<<dim3(256, 2),   dim3(256), 0, stream>>>(Wb, bv, wemb, sent1, sent2, chart);
  k_scan  <<<dim3(BSZ, 2),   dim3(256), 0, stream>>>(Ubw, bv, eu, unorm, ops1, ops2, chart);
  k_mlp1  <<<dim3(BSZ),      dim3(256), 0, stream>>>(chart, w1, b1, inter);
  k_mlp2  <<<dim3(1),        dim3(128), 0, stream>>>(inter, w2, b2, out);
}
